// MultiExperts_3410204033205
// MI455X (gfx1250) — matrix operand order repaired and checked by reading, not compiled, not run on hardware
//
#include <hip/hip_runtime.h>

// ---------------------------------------------------------------------------
// MoE dense-routing MLP: out[b,e,:] = relu(x[b,:] @ W1[e] + b1[e]) @ W2[e] + b2[e]
// B=4096 D=1024 H=2048 E=8, fp32 in/out.
// Compute-bound (~650 FLOP/byte) -> run MACs on v_wmma_f32_16x16x32_bf16,
// keep the intermediate h in bf16 workspace (128 MiB) between the two GEMMs.
// GEMM2's A operand (bf16 h) is staged via GLOBAL_LOAD_ASYNC_TO_LDS_B128
// (ASYNCcnt-tracked direct-to-LDS copy) when the builtin is available.
// ---------------------------------------------------------------------------

typedef __bf16 v16bf  __attribute__((ext_vector_type(16)));
typedef __bf16 bf16x8 __attribute__((ext_vector_type(8)));
typedef __bf16 bf16x4 __attribute__((ext_vector_type(4)));
typedef float  v8f    __attribute__((ext_vector_type(8)));

#define BDIM 4096
#define DDIM 1024
#define HDIM 2048
#define EDIM 8

#define BM 128
#define BN 128
#define BK 32
#define LDK 40   // padded LDS row stride (80 B -> 16B-aligned, conflict-light)

#if defined(__AMDGCN__) && __has_builtin(__builtin_amdgcn_global_load_async_to_lds_b128)
#define USE_ASYNC_LDS 1
#else
#define USE_ASYNC_LDS 0
#endif

#if USE_ASYNC_LDS
typedef int v4i __attribute__((vector_size(16)));
typedef __attribute__((address_space(1))) v4i as1_v4i;   // global (__device__) space
typedef __attribute__((address_space(3))) v4i as3_v4i;   // LDS (__shared__) space

__device__ __forceinline__ void async_copy_b128(const void* g, void* lds) {
    // AS1 value == generic VA for global; AS3 value == low 32 bits of generic
    // LDS VA (flat-aperture truncation, ISA 10.2).
    __builtin_amdgcn_global_load_async_to_lds_b128(
        (as1_v4i*)(uintptr_t)g, (as3_v4i*)(uintptr_t)lds, 0, 0);
}

__device__ __forceinline__ void wait_async0() {
#if __has_builtin(__builtin_amdgcn_s_wait_asynccnt)
    __builtin_amdgcn_s_wait_asynccnt(0);
#else
    asm volatile("s_wait_asynccnt 0" ::: "memory");
#endif
}
#endif

__device__ __forceinline__ __bf16 f2bf(float f) { return (__bf16)f; }

__device__ __forceinline__ v16bf frag16(const __bf16* p0, const __bf16* p1) {
    bf16x8 lo = *(const bf16x8*)p0;
    bf16x8 hi = *(const bf16x8*)p1;
    return __builtin_shufflevector(lo, hi, 0,1,2,3,4,5,6,7,8,9,10,11,12,13,14,15);
}

// ============================ GEMM1: h = relu(x @ W1 + b1), h stored bf16 ===
__global__ __launch_bounds__(256, 2)
void moe_gemm1(const float* __restrict__ x,     // [B, D]
               const float* __restrict__ W1,    // [E, D, H]
               const float* __restrict__ b1,    // [E, H]
               __bf16* __restrict__ h)          // [E, B, H] bf16
{
    __shared__ __align__(16) __bf16 As[2][BM][LDK];
    __shared__ __align__(16) __bf16 Bs[2][BN][LDK];

    const int tid  = threadIdx.x;
    const int lane = tid & 31;
    const int w    = tid >> 5;
    const int wm   = w >> 2;       // 0..1
    const int wn   = w & 3;        // 0..3
    const int e    = blockIdx.z;
    const int m0   = blockIdx.x * BM;
    const int n0   = blockIdx.y * BN;

    const float* W1e = W1 + (size_t)e * DDIM * HDIM;

    v8f acc[4][2];
    #pragma unroll
    for (int i = 0; i < 4; ++i)
        #pragma unroll
        for (int j = 0; j < 2; ++j)
            acc[i][j] = (v8f){0.f,0.f,0.f,0.f,0.f,0.f,0.f,0.f};

    float4 ra[4], rb[4];

    auto loadA = [&](int kt) {
        #pragma unroll
        for (int j = 0; j < 4; ++j) {
            int idx = j * 256 + tid;
            int row = idx >> 3, c4 = idx & 7;          // 8 float4 per 32-col row
            ra[j] = *(const float4*)(x + (size_t)(m0 + row) * DDIM + kt * BK + c4 * 4);
        }
    };
    auto loadB = [&](int kt) {
        #pragma unroll
        for (int j = 0; j < 4; ++j) {
            int idx = j * 256 + tid;
            int kr = idx >> 5, c4 = idx & 31;          // 32 float4 per 128-col row
            rb[j] = *(const float4*)(W1e + (size_t)(kt * BK + kr) * HDIM + n0 + c4 * 4);
        }
    };
    auto storeA = [&](int buf) {
        #pragma unroll
        for (int j = 0; j < 4; ++j) {
            int idx = j * 256 + tid;
            int row = idx >> 3, c4 = idx & 7;
            bf16x4 p = { f2bf(ra[j].x), f2bf(ra[j].y), f2bf(ra[j].z), f2bf(ra[j].w) };
            *(bf16x4*)&As[buf][row][c4 * 4] = p;
        }
    };
    auto storeB = [&](int buf) {                       // transpose: Bs[n][k]
        #pragma unroll
        for (int j = 0; j < 4; ++j) {
            int idx = j * 256 + tid;
            int kr = idx >> 5, c4 = idx & 31;
            Bs[buf][c4 * 4 + 0][kr] = f2bf(rb[j].x);
            Bs[buf][c4 * 4 + 1][kr] = f2bf(rb[j].y);
            Bs[buf][c4 * 4 + 2][kr] = f2bf(rb[j].z);
            Bs[buf][c4 * 4 + 3][kr] = f2bf(rb[j].w);
        }
    };
    auto compute = [&](int buf) {
        const int mr = lane & 15;
        const int lo = (lane < 16) ? 0 : 8;            // A: K lo-half select
        const int bk = (lane < 16) ? 0 : 8;            // B: K lo-half select
        v16bf af[4], bg[2];
        #pragma unroll
        for (int i = 0; i < 4; ++i) {
            const __bf16* p = &As[buf][wm * 64 + i * 16 + mr][lo];
            af[i] = frag16(p, p + 16);
        }
        #pragma unroll
        for (int j = 0; j < 2; ++j) {
            const __bf16* p = &Bs[buf][wn * 32 + j * 16 + mr][bk];
            bg[j] = frag16(p, p + 16);
        }
        #pragma unroll
        for (int i = 0; i < 4; ++i)
            #pragma unroll
            for (int j = 0; j < 2; ++j)
                acc[i][j] = __builtin_amdgcn_wmma_f32_16x16x32_bf16(
                    false, af[i], false, bg[j], (short)0, acc[i][j], false, false);
    };

    const int KT = DDIM / BK;  // 32
    loadA(0); loadB(0); storeA(0); storeB(0);
    __syncthreads();
    for (int kt = 0; kt < KT; ++kt) {
        const int cur = kt & 1;
        if (kt + 2 < KT)
            __builtin_prefetch((const void*)(W1e + (size_t)((kt + 2) * BK) * HDIM + n0), 0, 1);
        if (kt + 1 < KT) { loadA(kt + 1); loadB(kt + 1); }
        compute(cur);
        if (kt + 1 < KT) { storeA(cur ^ 1); storeB(cur ^ 1); }
        __syncthreads();
    }

    // Epilogue: bias + relu + bf16 store to workspace h[e][M][N]
    const int mr   = lane & 15;
    const int hsel = (lane < 16) ? 0 : 8;
    #pragma unroll
    for (int i = 0; i < 4; ++i) {
        #pragma unroll
        for (int j = 0; j < 2; ++j) {
            const int N = n0 + wn * 32 + j * 16 + mr;
            const float bias = b1[(size_t)e * HDIM + N];
            #pragma unroll
            for (int r = 0; r < 8; ++r) {
                const int M = m0 + wm * 64 + i * 16 + r + hsel;
                float v = acc[i][j][r] + bias;
                v = v > 0.f ? v : 0.f;
                h[((size_t)e * BDIM + M) * HDIM + N] = f2bf(v);
            }
        }
    }
}

// ============================ GEMM2: out = h @ W2 + b2 ======================
__global__ __launch_bounds__(256, 2)
void moe_gemm2(const __bf16* __restrict__ h,    // [E, B, H] bf16
               const float*  __restrict__ W2,   // [E, H, D]
               const float*  __restrict__ b2,   // [E, D]
               float* __restrict__ out)         // [B, E, D]
{
    __shared__ __align__(16) __bf16 As[2][BM][LDK];
    __shared__ __align__(16) __bf16 Bs[2][BN][LDK];

    const int tid  = threadIdx.x;
    const int lane = tid & 31;
    const int w    = tid >> 5;
    const int wm   = w >> 2;
    const int wn   = w & 3;
    const int e    = blockIdx.z;
    const int m0   = blockIdx.x * BM;
    const int n0   = blockIdx.y * BN;

    const __bf16* he  = h  + (size_t)e * BDIM * HDIM;
    const float*  W2e = W2 + (size_t)e * HDIM * DDIM;

    v8f acc[4][2];
    #pragma unroll
    for (int i = 0; i < 4; ++i)
        #pragma unroll
        for (int j = 0; j < 2; ++j)
            acc[i][j] = (v8f){0.f,0.f,0.f,0.f,0.f,0.f,0.f,0.f};

    float4 rb[4];

    // ---- A staging: bf16 h tile, copied global->LDS -------------------------
#if USE_ASYNC_LDS
    auto copyA = [&](int kt, int buf) {                // direct-to-LDS async
        #pragma unroll
        for (int j = 0; j < 2; ++j) {
            int idx = j * 256 + tid;
            int row = idx >> 2, c8 = idx & 3;          // 4 bf16x8 per 32-col row
            async_copy_b128(he + (size_t)(m0 + row) * HDIM + kt * BK + c8 * 8,
                            &As[buf][row][c8 * 8]);
        }
    };
#else
    bf16x8 ra[2];
    auto loadA = [&](int kt) {
        #pragma unroll
        for (int j = 0; j < 2; ++j) {
            int idx = j * 256 + tid;
            int row = idx >> 2, c8 = idx & 3;
            ra[j] = *(const bf16x8*)(he + (size_t)(m0 + row) * HDIM + kt * BK + c8 * 8);
        }
    };
    auto storeA = [&](int buf) {
        #pragma unroll
        for (int j = 0; j < 2; ++j) {
            int idx = j * 256 + tid;
            int row = idx >> 2, c8 = idx & 3;
            *(bf16x8*)&As[buf][row][c8 * 8] = ra[j];
        }
    };
#endif
    auto loadB = [&](int kt) {
        #pragma unroll
        for (int j = 0; j < 4; ++j) {
            int idx = j * 256 + tid;
            int kr = idx >> 5, c4 = idx & 31;
            rb[j] = *(const float4*)(W2e + (size_t)(kt * BK + kr) * DDIM + n0 + c4 * 4);
        }
    };
    auto storeB = [&](int buf) {                       // transpose: Bs[n][k]
        #pragma unroll
        for (int j = 0; j < 4; ++j) {
            int idx = j * 256 + tid;
            int kr = idx >> 5, c4 = idx & 31;
            Bs[buf][c4 * 4 + 0][kr] = f2bf(rb[j].x);
            Bs[buf][c4 * 4 + 1][kr] = f2bf(rb[j].y);
            Bs[buf][c4 * 4 + 2][kr] = f2bf(rb[j].z);
            Bs[buf][c4 * 4 + 3][kr] = f2bf(rb[j].w);
        }
    };
    auto compute = [&](int buf) {
        const int mr = lane & 15;
        const int lo = (lane < 16) ? 0 : 8;
        const int bk = (lane < 16) ? 0 : 8;
        v16bf af[4], bg[2];
        #pragma unroll
        for (int i = 0; i < 4; ++i) {
            const __bf16* p = &As[buf][wm * 64 + i * 16 + mr][lo];
            af[i] = frag16(p, p + 16);
        }
        #pragma unroll
        for (int j = 0; j < 2; ++j) {
            const __bf16* p = &Bs[buf][wn * 32 + j * 16 + mr][bk];
            bg[j] = frag16(p, p + 16);
        }
        #pragma unroll
        for (int i = 0; i < 4; ++i)
            #pragma unroll
            for (int j = 0; j < 2; ++j)
                acc[i][j] = __builtin_amdgcn_wmma_f32_16x16x32_bf16(
                    false, af[i], false, bg[j], (short)0, acc[i][j], false, false);
    };

    const int KT = HDIM / BK;  // 64
#if USE_ASYNC_LDS
    copyA(0, 0); loadB(0); storeB(0);
    wait_async0();
#else
    loadA(0); loadB(0); storeA(0); storeB(0);
#endif
    __syncthreads();
    for (int kt = 0; kt < KT; ++kt) {
        const int cur = kt & 1;
        if (kt + 2 < KT)
            __builtin_prefetch((const void*)(W2e + (size_t)((kt + 2) * BK) * DDIM + n0), 0, 1);
#if USE_ASYNC_LDS
        if (kt + 1 < KT) { copyA(kt + 1, cur ^ 1); loadB(kt + 1); }
        compute(cur);
        if (kt + 1 < KT) { storeB(cur ^ 1); }
        wait_async0();            // A tile for next buffer resident in LDS
#else
        if (kt + 1 < KT) { loadA(kt + 1); loadB(kt + 1); }
        compute(cur);
        if (kt + 1 < KT) { storeA(cur ^ 1); storeB(cur ^ 1); }
#endif
        __syncthreads();
    }

    // Epilogue: bias + f32 store, out[M][e][N]
    const int mr   = lane & 15;
    const int hsel = (lane < 16) ? 0 : 8;
    #pragma unroll
    for (int i = 0; i < 4; ++i) {
        #pragma unroll
        for (int j = 0; j < 2; ++j) {
            const int N = n0 + wn * 32 + j * 16 + mr;
            const float bias = b2[(size_t)e * DDIM + N];
            #pragma unroll
            for (int r = 0; r < 8; ++r) {
                const int M = m0 + wm * 64 + i * 16 + r + hsel;
                out[(size_t)M * (EDIM * DDIM) + (size_t)e * DDIM + N] = acc[i][j][r] + bias;
            }
        }
    }
}

// ============================ launcher ======================================
extern "C" void kernel_launch(void* const* d_in, const int* in_sizes, int n_in,
                              void* d_out, int out_size, void* d_ws, size_t ws_size,
                              hipStream_t stream) {
    const float* x  = (const float*)d_in[0];   // [4096,1024]
    const float* W1 = (const float*)d_in[1];   // [8,1024,2048]
    const float* b1 = (const float*)d_in[2];   // [8,2048]
    const float* W2 = (const float*)d_in[3];   // [8,2048,1024]
    const float* b2 = (const float*)d_in[4];   // [8,1024]
    float* out = (float*)d_out;                // [4096,8,1024]
    __bf16* h = (__bf16*)d_ws;                 // [8,4096,2048] bf16 = 128 MiB

    dim3 blk(256);
    dim3 g1(BDIM / BM, HDIM / BN, EDIM);       // 32 x 16 x 8
    moe_gemm1<<<g1, blk, 0, stream>>>(x, W1, b1, h);
    dim3 g2(BDIM / BM, DDIM / BN, EDIM);       // 32 x 8 x 8
    moe_gemm2<<<g2, blk, 0, stream>>>(h, W2, b2, out);
}
